// SeqVLADModule_30056181137436
// MI455X (gfx1250) — matrix operand order repaired and checked by reading, not compiled, not run on hardware
//
#include <hip/hip_runtime.h>
#include <math.h>

// ---------------- sizes ----------------
#define BSZ   32
#define TT    8
#define KK    64
#define DD    512
#define CIN   1536
#define HW    64
#define NIMG  (BSZ*TT)          // 256
#define PP    (NIMG*HW)         // 16384
#define KIM   (KK*9)            // 576  (im2col K-dim)
#define MH    (BSZ*HW)          // 2048 (GRU conv rows)
#define TK    (TT*KK)           // 512

typedef __attribute__((ext_vector_type(16))) __bf16 v16bf;
typedef __attribute__((ext_vector_type(8)))  float  v8f;

union FragBF { v16bf v; unsigned int u[8]; };
union FragF  { v8f   v; float f[8]; };

__device__ __forceinline__ unsigned short f2bf(float f) {
  unsigned int u = __float_as_uint(f);
  u += 0x7FFFu + ((u >> 16) & 1u);          // round-to-nearest-even
  return (unsigned short)(u >> 16);
}
__device__ __forceinline__ float bf2f(unsigned short h) {
  return __uint_as_float(((unsigned int)h) << 16);
}
__device__ __forceinline__ float wred_sum(float v) {
  for (int o = 16; o; o >>= 1) v += __shfl_xor(v, o, 32);
  return v;
}
__device__ __forceinline__ float wred_max(float v) {
  for (int o = 16; o; o >>= 1) v = fmaxf(v, __shfl_xor(v, o, 32));
  return v;
}
__device__ __forceinline__ float block_sum(float v) {
  __shared__ float s[8];
  int lane = threadIdx.x & 31, wid = threadIdx.x >> 5;
  float w = wred_sum(v);
  if (lane == 0) s[wid] = w;
  __syncthreads();
  float t;
  if (threadIdx.x == 0) { t = 0.f; for (int i = 0; i < 8; i++) t += s[i]; s[0] = t; }
  __syncthreads();
  t = s[0];
  __syncthreads();
  return t;
}

// ---------------- generic WMMA GEMM ----------------
// out[m,n] = act( sum_k A[m,k]*B'[k,n] + bias[n] + addend )
// BSRC=0: Bm = W row-major (N x K), ldb = K
// BSRC=1: Bm = B row-major (K x N), ldb = N      (per-64-row image B-base stride)
// SM=0: out row-major (M x N);  SM=1: NCHW scatter out[(m/64), n, m%64]
// ACT: 0 none, 1 relu, 2 sigmoid, 3 tanh
template <int BSRC, int SM, int ACT>
__global__ __launch_bounds__(256) void gemm_k(
    const unsigned short* __restrict__ A, int lda,
    const unsigned short* __restrict__ Bm, int ldb, long long bImgStride,
    float* __restrict__ out, int M, int N, int Kd,
    const float* __restrict__ bias, const float* __restrict__ addend,
    unsigned short* __restrict__ bf16copy)
{
  int wid = blockIdx.x * (blockDim.x >> 5) + (threadIdx.x >> 5);
  int mtiles = M >> 4, ntiles = N >> 6;
  if (wid >= mtiles * ntiles) return;
  int mt = wid % mtiles, nt = wid / mtiles;
  int lane = threadIdx.x & 31;

  const unsigned short* Beff = Bm + (long long)(mt >> 2) * bImgStride; // (mt*16)/64
  int mrow = mt * 16 + (lane & 15);
  const unsigned short* Arow = A + (size_t)mrow * lda;

  FragF acc[4];
  v8f zero = {0.f, 0.f, 0.f, 0.f, 0.f, 0.f, 0.f, 0.f};
#pragma unroll
  for (int j = 0; j < 4; j++) acc[j].v = zero;

  for (int kk = 0; kk < Kd; kk += 32) {
    FragBF fa;
    {
      // A 16x32 bf16 layout: lanes 0-15 -> K kk..kk+15; lanes 16-31 -> kk+16..kk+31
      int k0 = kk + ((lane >> 4) << 4);
      uint4 x0 = *(const uint4*)(Arow + k0);
      uint4 x1 = *(const uint4*)(Arow + k0 + 8);
      fa.u[0] = x0.x; fa.u[1] = x0.y; fa.u[2] = x0.z; fa.u[3] = x0.w;
      fa.u[4] = x1.x; fa.u[5] = x1.y; fa.u[6] = x1.z; fa.u[7] = x1.w;
    }
#pragma unroll
    for (int j = 0; j < 4; j++) {
      FragBF fb;
      int n = nt * 64 + j * 16 + (lane & 15);
      int k0 = kk + ((lane >> 4) << 4); // lanes 0-15: K kk..kk+15; 16-31: kk+16..kk+31
      if (BSRC == 0) {
        const unsigned short* p = Beff + (size_t)n * ldb + k0;
        uint4 x0 = *(const uint4*)(p);
        uint4 x1 = *(const uint4*)(p + 8);
        fb.u[0] = x0.x; fb.u[1] = x0.y; fb.u[2] = x0.z; fb.u[3] = x0.w;
        fb.u[4] = x1.x; fb.u[5] = x1.y; fb.u[6] = x1.z; fb.u[7] = x1.w;
      } else {
#pragma unroll
        for (int i = 0; i < 8; i++) {
          unsigned int lo = Beff[(size_t)(k0 + 2 * i) * ldb + n];
          unsigned int hi = Beff[(size_t)(k0 + 2 * i + 1) * ldb + n];
          fb.u[i] = lo | (hi << 16);
        }
      }
      acc[j].v = __builtin_amdgcn_wmma_f32_16x16x32_bf16(
          false, fa.v, false, fb.v, (short)0, acc[j].v, false, false);
    }
  }

  int nl = lane & 15;
  int mbase = (lane >> 4) << 3;
#pragma unroll
  for (int j = 0; j < 4; j++) {
    int n = nt * 64 + j * 16 + nl;
    float bv = bias ? bias[n] : 0.f;
#pragma unroll
    for (int i = 0; i < 8; i++) {
      int m = mt * 16 + mbase + i;
      float v = acc[j].f[i] + bv;
      size_t idx;
      if (SM == 0) idx = (size_t)m * N + n;
      else { int img = m >> 6; int pos = m & 63; idx = (size_t)img * ((size_t)N * 64) + (size_t)n * 64 + pos; }
      if (addend) v += addend[idx];
      if (ACT == 1) v = fmaxf(v, 0.f);
      else if (ACT == 2) v = 1.f / (1.f + __expf(-v));
      else if (ACT == 3) v = tanhf(v);
      out[idx] = v;
      if (SM == 0 && bf16copy) bf16copy[idx] = f2bf(v);
    }
  }
}

// ---------------- small kernels ----------------
__global__ void cast_bf16_k(const float* __restrict__ s, unsigned short* __restrict__ d, int n) {
  int i = blockIdx.x * blockDim.x + threadIdx.x;
  if (i < n) d[i] = f2bf(s[i]);
}
__global__ void zerof_k(float* p, int n) {
  int i = blockIdx.x * blockDim.x + threadIdx.x;
  if (i < n) p[i] = 0.f;
}
// x (N,CIN,8,8) fp32 -> xb (P, CIN) bf16
__global__ void xbpack_k(const float* __restrict__ x, unsigned short* __restrict__ xb) {
  int i = blockIdx.x * blockDim.x + threadIdx.x;
  if (i >= PP * CIN) return;
  int p = i / CIN, c = i - p * CIN;
  int n = p >> 6, hw = p & 63;
  xb[i] = f2bf(x[((size_t)n * CIN + c) * 64 + hw]);
}
// in (Nimg,64,8,8) fp32 -> col (Nimg*64, 576) bf16
__global__ void im2col_k(const float* __restrict__ in, unsigned short* __restrict__ col, int total) {
  int i = blockIdx.x * blockDim.x + threadIdx.x;
  if (i >= total) return;
  int q = i % KIM, rem = i / KIM;
  int pos = rem & 63, n = rem >> 6;
  int c = q / 9, tap = q - c * 9;
  int dy = tap / 3 - 1, dx = tap - (tap / 3) * 3 - 1;
  int y = (pos >> 3) + dy, xx = (pos & 7) + dx;
  float v = (y >= 0 && y < 8 && xx >= 0 && xx < 8) ? in[((size_t)n * 64 + c) * 64 + y * 8 + xx] : 0.f;
  col[i] = f2bf(v);
}
// per (b,t,k): mean & max over hw of relu(attv[b*8+t] + ah[b])
__global__ void ereduce_k(const float* __restrict__ attv, const float* __restrict__ ah,
                          float* __restrict__ emean, float* __restrict__ emax) {
  int w = blockIdx.x * (blockDim.x >> 5) + (threadIdx.x >> 5);
  if (w >= BSZ * TK) return;
  int lane = threadIdx.x & 31;
  int b = w >> 9, t = (w >> 6) & 7, k = w & 63;
  const float* av = attv + ((size_t)(b * 8 + t) * 64 + k) * 64;
  const float* ab = ah + ((size_t)b * 64 + k) * 64;
  float v1 = fmaxf(av[lane] + ab[lane], 0.f);
  float v2 = fmaxf(av[lane + 32] + ab[lane + 32], 0.f);
  float s = wred_sum(v1 + v2);
  float m = wred_max(fmaxf(v1, v2));
  if (lane == 0) {
    emean[(size_t)b * TK + t * 64 + k] = s * (1.f / 64.f);
    emax[(size_t)b * TK + t * 64 + k] = m;
  }
}
// shared-MLP channel attention + softmax over T -> alphas (B, T*K)
__global__ __launch_bounds__(256) void gate_k(const float* __restrict__ emean, const float* __restrict__ emax,
                                              const float* __restrict__ fc1, const float* __restrict__ fc2,
                                              float* __restrict__ alphas) {
  int b = blockIdx.x, tid = threadIdx.x;
  __shared__ float h1[32];
  __shared__ float ex[TK];
  if (tid < 32) {
    const float* w = fc1 + tid * TK;
    const float* em = emean + (size_t)b * TK;
    const float* ea = emax + (size_t)b * TK;
    float sm = 0.f, sx = 0.f;
    for (int q = 0; q < TK; q++) { sm += w[q] * em[q]; sx += w[q] * ea[q]; }
    h1[tid] = fmaxf(sm, 0.f) + fmaxf(sx, 0.f); // fc2 is linear: sum post-relu hiddens
  }
  __syncthreads();
  for (int j = tid; j < TK; j += 256) {
    float s = 0.f;
    for (int i = 0; i < 32; i++) s += fc2[j * 32 + i] * h1[i];
    ex[j] = __expf(tanhf(s));
  }
  __syncthreads();
  for (int k = tid; k < KK; k += 256) {
    float den = 0.f;
    for (int t = 0; t < TT; t++) den += ex[t * 64 + k];
    den += (den == 0.f) ? 1.f : 0.f;
    float inv = 1.f / den;
    for (int t = 0; t < TT; t++) alphas[(size_t)b * TK + t * 64 + k] = ex[t * 64 + k] * inv;
  }
}
// wxt0[b,k,hw] = sum_t alphas[b,t,k]*wxpb[b*8+t,k,hw]
__global__ void wxt0_k(const float* __restrict__ alphas, const float* __restrict__ wxpb,
                       float* __restrict__ wxt0) {
  int i = blockIdx.x * blockDim.x + threadIdx.x;
  if (i >= BSZ * KK * HW) return;
  int b = i >> 12, k = (i >> 6) & 63, hw = i & 63;
  float s = 0.f;
  for (int t = 0; t < TT; t++)
    s += alphas[(size_t)b * TK + t * 64 + k] * wxpb[((size_t)(b * 8 + t) * 64 + k) * 64 + hw];
  wxt0[i] = s;
}
__global__ void mul_k(const float* a, const float* b, float* c, int n) {
  int i = blockIdx.x * blockDim.x + threadIdx.x;
  if (i < n) c[i] = a[i] * b[i];
}
// h = (1-z)*hh + z*h ; also scatter bf16 into assigns (b,k, t*64+s)
__global__ void hupdate_k(const float* __restrict__ z, const float* __restrict__ hh,
                          float* __restrict__ h, unsigned short* __restrict__ assigns, int t) {
  int i = blockIdx.x * blockDim.x + threadIdx.x;
  if (i >= BSZ * KK * HW) return;
  float zz = z[i];
  float hn = (1.f - zz) * hh[i] + zz * h[i];
  h[i] = hn;
  int b = i >> 12, k = (i >> 6) & 63, s = i & 63;
  assigns[((size_t)b * 64 + k) * TK + t * 64 + s] = f2bf(hn);
}
__global__ void asum_k(const unsigned short* __restrict__ assigns, float* __restrict__ asum) {
  int w = blockIdx.x * (blockDim.x >> 5) + (threadIdx.x >> 5);
  if (w >= BSZ * KK) return;
  int lane = threadIdx.x & 31;
  float s = 0.f;
  for (int j = 0; j < 16; j++) s += bf2f(assigns[(size_t)w * TK + lane + 32 * j]);
  s = wred_sum(s);
  if (lane == 0) asum[w] = s;
}
// subtract a_sum*centers, intra-normalize over D (per b,k), in place
__global__ __launch_bounds__(256) void vpost1_k(float* __restrict__ vlad, const float* __restrict__ asum,
                                                const float* __restrict__ centers) {
  int row = blockIdx.x, tid = threadIdx.x;
  int k = row & 63;
  float a = asum[row];
  float v0 = vlad[(size_t)row * DD + tid]       - a * centers[(size_t)k * DD + tid];
  float v1 = vlad[(size_t)row * DD + tid + 256] - a * centers[(size_t)k * DD + tid + 256];
  float ss = block_sum(v0 * v0 + v1 * v1);
  float inv = 1.f / fmaxf(sqrtf(ss), 1e-12f);
  vlad[(size_t)row * DD + tid] = v0 * inv;
  vlad[(size_t)row * DD + tid + 256] = v1 * inv;
}
// final L2 over K*D per batch -> out
__global__ __launch_bounds__(256) void vpost2_k(const float* __restrict__ vlad, float* __restrict__ out) {
  int b = blockIdx.x, tid = threadIdx.x;
  float ss = 0.f;
  for (int i = tid; i < KK * DD; i += 256) { float v = vlad[(size_t)b * KK * DD + i]; ss += v * v; }
  ss = block_sum(ss);
  float inv = 1.f / fmaxf(sqrtf(ss), 1e-12f);
  for (int i = tid; i < KK * DD; i += 256)
    out[(size_t)b * KK * DD + i] = vlad[(size_t)b * KK * DD + i] * inv;
}

// ---------------- host ----------------
static inline int cdiv(long long a, int b) { return (int)((a + b - 1) / b); }

extern "C" void kernel_launch(void* const* d_in, const int* in_sizes, int n_in,
                              void* d_out, int out_size, void* d_ws, size_t ws_size,
                              hipStream_t stream) {
  const float* x      = (const float*)d_in[0];
  const float* redu_w = (const float*)d_in[1];
  const float* redu_b = (const float*)d_in[2];
  const float* w_x    = (const float*)d_in[3];
  const float* att_x  = (const float*)d_in[4];
  const float* att_hw = (const float*)d_in[5];
  const float* att_b  = (const float*)d_in[6];
  const float* sh_w   = (const float*)d_in[7];
  const float* sh_b   = (const float*)d_in[8];
  const float* U_r    = (const float*)d_in[9];
  const float* U_z    = (const float*)d_in[10];
  const float* U_h    = (const float*)d_in[11];
  const float* centers= (const float*)d_in[12];
  const float* fc1    = (const float*)d_in[13];
  const float* fc2    = (const float*)d_in[14];
  float* out = (float*)d_out;

  char* base = (char*)d_ws;
  size_t off = 0;
  auto alloc = [&](size_t bytes) -> char* {
    char* p = base + off;
    off += (bytes + 255) & ~(size_t)255;
    return p;
  };
  unsigned short* xb    = (unsigned short*)alloc((size_t)PP * CIN * 2);   // dies after GEMM1
  unsigned short* rwb   = (unsigned short*)alloc((size_t)DD * CIN * 2);
  unsigned short* wxb   = (unsigned short*)alloc((size_t)KK * DD * 2);
  unsigned short* cw    = (unsigned short*)alloc((size_t)6 * KK * KIM * 2);
  float* xt    = (float*)alloc((size_t)PP * DD * 4);
  unsigned short* xtb = (unsigned short*)alloc((size_t)PP * DD * 2);
  float* wxpb  = (float*)alloc((size_t)NIMG * KK * HW * 4);
  float* attv  = (float*)alloc((size_t)NIMG * KK * HW * 4);
  float* ah    = (float*)alloc((size_t)BSZ * KK * HW * 4);
  float* wxt0  = (float*)alloc((size_t)BSZ * KK * HW * 4);
  float* wxt   = (float*)alloc((size_t)BSZ * KK * HW * 4);
  float* zb    = (float*)alloc((size_t)BSZ * KK * HW * 4);
  float* rb    = (float*)alloc((size_t)BSZ * KK * HW * 4);
  float* rh    = (float*)alloc((size_t)BSZ * KK * HW * 4);
  float* hh    = (float*)alloc((size_t)BSZ * KK * HW * 4);
  float* hbuf  = (float*)alloc((size_t)BSZ * KK * HW * 4);
  float* emean = (float*)alloc((size_t)BSZ * TK * 4);
  float* emax  = (float*)alloc((size_t)BSZ * TK * 4);
  float* alphas= (float*)alloc((size_t)BSZ * TK * 4);
  unsigned short* assigns = (unsigned short*)alloc((size_t)BSZ * KK * TK * 2);
  float* asum  = (float*)alloc((size_t)BSZ * KK * 4);
  float* vlad  = (float*)alloc((size_t)BSZ * KK * DD * 4);
  // big im2col buffer aliased into xb (xb is dead after GEMM1; col fits in 50MB)
  unsigned short* colbig  = xb;                                    // NIMG*64 x 576
  unsigned short* colh    = colbig;                                // 2048 x 576
  unsigned short* colmisc = colbig + (size_t)MH * KIM;             // 2048 x 576

  unsigned short* cwA = cw + 0 * (size_t)KK * KIM; // att_x
  unsigned short* cwH = cw + 1 * (size_t)KK * KIM; // att_h_w
  unsigned short* cwS = cw + 2 * (size_t)KK * KIM; // share_w
  unsigned short* cwR = cw + 3 * (size_t)KK * KIM; // U_r
  unsigned short* cwZ = cw + 4 * (size_t)KK * KIM; // U_z
  unsigned short* cwU = cw + 5 * (size_t)KK * KIM; // U_h

  // weight packing to bf16
  cast_bf16_k<<<cdiv(DD * CIN, 256), 256, 0, stream>>>(redu_w, rwb, DD * CIN);
  cast_bf16_k<<<cdiv(KK * DD, 256), 256, 0, stream>>>(w_x, wxb, KK * DD);
  cast_bf16_k<<<cdiv(KK * KIM, 256), 256, 0, stream>>>(att_x, cwA, KK * KIM);
  cast_bf16_k<<<cdiv(KK * KIM, 256), 256, 0, stream>>>(att_hw, cwH, KK * KIM);
  cast_bf16_k<<<cdiv(KK * KIM, 256), 256, 0, stream>>>(sh_w, cwS, KK * KIM);
  cast_bf16_k<<<cdiv(KK * KIM, 256), 256, 0, stream>>>(U_r, cwR, KK * KIM);
  cast_bf16_k<<<cdiv(KK * KIM, 256), 256, 0, stream>>>(U_z, cwZ, KK * KIM);
  cast_bf16_k<<<cdiv(KK * KIM, 256), 256, 0, stream>>>(U_h, cwU, KK * KIM);
  xbpack_k<<<cdiv((long long)PP * CIN, 256), 256, 0, stream>>>(x, xb);
  zerof_k<<<cdiv(BSZ * KK * HW, 256), 256, 0, stream>>>(hbuf, BSZ * KK * HW);

  // GEMM1: xt = xb(P x CIN) * redu_w^T + redu_b  (also bf16 copy xtb)
  gemm_k<0, 0, 0><<<cdiv((PP / 16) * (DD / 64), 8), 256, 0, stream>>>(
      xb, CIN, rwb, CIN, 0, xt, PP, DD, CIN, redu_b, nullptr, xtb);
  // GEMM2: wxpb(NCHW) = xt * w_x^T
  gemm_k<0, 1, 0><<<cdiv((PP / 16) * (KK / 64), 8), 256, 0, stream>>>(
      xtb, DD, wxb, DD, 0, wxpb, PP, KK, DD, nullptr, nullptr, nullptr);
  // att_v = conv3x3(wxpb, att_x)
  im2col_k<<<cdiv((long long)NIMG * HW * KIM, 256), 256, 0, stream>>>(wxpb, colbig, NIMG * HW * KIM);
  gemm_k<0, 1, 0><<<cdiv((PP / 16), 8), 256, 0, stream>>>(
      colbig, KIM, cwA, KIM, 0, attv, PP, KK, KIM, nullptr, nullptr, nullptr);

  const int nEl = BSZ * KK * HW;
  const int convBlocks = cdiv(MH / 16, 8); // 16
  for (int t = 0; t < TT; t++) {
    im2col_k<<<cdiv((long long)MH * KIM, 256), 256, 0, stream>>>(hbuf, colh, MH * KIM);
    // ah = conv(h, att_h_w) + att_b
    gemm_k<0, 1, 0><<<convBlocks, 256, 0, stream>>>(
        colh, KIM, cwH, KIM, 0, ah, MH, KK, KIM, att_b, nullptr, nullptr);
    ereduce_k<<<cdiv(BSZ * TK, 8), 256, 0, stream>>>(attv, ah, emean, emax);
    gate_k<<<BSZ, 256, 0, stream>>>(emean, emax, fc1, fc2, alphas);
    wxt0_k<<<cdiv(nEl, 256), 256, 0, stream>>>(alphas, wxpb, wxt0);
    // wxt = conv(wxt0, share_w) + share_b
    im2col_k<<<cdiv((long long)MH * KIM, 256), 256, 0, stream>>>(wxt0, colmisc, MH * KIM);
    gemm_k<0, 1, 0><<<convBlocks, 256, 0, stream>>>(
        colmisc, KIM, cwS, KIM, 0, wxt, MH, KK, KIM, sh_b, nullptr, nullptr);
    // z = sigmoid(wxt + conv(h,U_z)); r = sigmoid(wxt + conv(h,U_r))
    gemm_k<0, 1, 2><<<convBlocks, 256, 0, stream>>>(
        colh, KIM, cwZ, KIM, 0, zb, MH, KK, KIM, nullptr, wxt, nullptr);
    gemm_k<0, 1, 2><<<convBlocks, 256, 0, stream>>>(
        colh, KIM, cwR, KIM, 0, rb, MH, KK, KIM, nullptr, wxt, nullptr);
    mul_k<<<cdiv(nEl, 256), 256, 0, stream>>>(rb, hbuf, rh, nEl);
    im2col_k<<<cdiv((long long)MH * KIM, 256), 256, 0, stream>>>(rh, colmisc, MH * KIM);
    // hh = tanh(wxt + conv(r*h, U_h))
    gemm_k<0, 1, 3><<<convBlocks, 256, 0, stream>>>(
        colmisc, KIM, cwU, KIM, 0, hh, MH, KK, KIM, nullptr, wxt, nullptr);
    hupdate_k<<<cdiv(nEl, 256), 256, 0, stream>>>(zb, hh, hbuf, assigns, t);
  }

  // VLAD: per-b GEMM assigns(64 x 512) * feats(512 x 512); feats rows are xt rows b*512..
  asum_k<<<cdiv(BSZ * KK, 8), 256, 0, stream>>>(assigns, asum);
  gemm_k<1, 0, 0><<<cdiv((BSZ * KK / 16) * (DD / 64), 8), 256, 0, stream>>>(
      assigns, TK, xtb, DD, (long long)TK * DD, vlad, BSZ * KK, DD, TK,
      nullptr, nullptr, nullptr);
  vpost1_k<<<BSZ * KK, 256, 0, stream>>>(vlad, asum, centers);
  vpost2_k<<<BSZ, 256, 0, stream>>>(vlad, out);
  (void)in_sizes; (void)n_in; (void)out_size; (void)ws_size;
}
